// ENLCA_27410481283656
// MI455X (gfx1250) — matrix operand order repaired and checked by reading, not compiled, not run on hardware
//
#include <hip/hip_runtime.h>
#include <hip/hip_bf16.h>

// ---------------------------------------------------------------------------
// ENLCA (Performer linear attention) for MI455X / gfx1250.
// All matmuls via v_wmma_f32_16x16x32_f16 (wave32 WMMA, f32 accumulate).
// LDS staging of f16 tiles via the Tensor Data Mover (tensor_load_to_lds).
// ---------------------------------------------------------------------------

typedef __attribute__((ext_vector_type(16))) _Float16 v16h;
typedef __attribute__((ext_vector_type(8)))  float    v8f;
typedef __attribute__((ext_vector_type(4)))  unsigned v4u;
typedef __attribute__((ext_vector_type(8)))  int      v8i;
typedef __attribute__((ext_vector_type(4)))  int      v4i;

#define NB 8
#define CC 128
#define LLEN 16384            // H*W
#define NL 131072             // NB*LLEN
#define K_SCALE   2.4494897427831781f   // sqrt(6)
#define RES_SCALE 0.1f
#define NORM_EPS  5e-5f
#define KERN_EPS  1e-4f
#define DNORM     0.35355339059327373f  // 64^-0.25
#define RATIO     0.08838834764831845f  // 128^-0.5

#define HAVE_TDM __has_builtin(__builtin_amdgcn_tensor_load_to_lds)

// ---------------------------------------------------------------------------
// Tensor Data Mover: 1-D contiguous f16 block copy global -> LDS.
// D# per CDNA5 ISA 8.3/8.4: group0 = {count=1, lds_addr, global_addr, type=2},
// group1 = {data_size=1(2B), tensor_dim0=n, tile_dim0=n, stride0=n}.
// Issue from one wave; descriptor operands are wave-uniform (SGPRs).
// ---------------------------------------------------------------------------
__device__ __forceinline__ void tdm_copy_f16(void* lds_dst, const void* gsrc,
                                             unsigned nelem /* <= 65535 */) {
#if HAVE_TDM
  unsigned long long ga = (unsigned long long)gsrc;
  v4u g0;
  g0[0] = 1u;                                               // count=1, user mode
  g0[1] = (unsigned)(unsigned long long)lds_dst;            // LDS byte address
  g0[2] = (unsigned)ga;                                     // global_addr[31:0]
  g0[3] = (unsigned)((ga >> 32) & 0x01FFFFFFull) | (2u << 30); // addr[56:32], type=2
  v8i g1;
  g1[0] = 0x00010000;                        // workgroup_mask=0, data_size=1 (2B)
  g1[1] = (int)((nelem & 0xFFFFu) << 16);    // tensor_dim0 lo16 (bits 63:48)
  g1[2] = (int)(nelem >> 16);                // tensor_dim0 hi16; tensor_dim1=0
  g1[3] = (int)((nelem & 0xFFFFu) << 16);    // tile_dim0 (bits 127:112)
  g1[4] = 0;                                 // tile_dim1=0 (unused), tile_dim2=0
  g1[5] = (int)nelem;                        // tensor_dim0_stride lo32
  g1[6] = 0;
  g1[7] = 0;
  v4i z = {0, 0, 0, 0};
#if __clang_major__ >= 23
  v8i z8 = {0, 0, 0, 0, 0, 0, 0, 0};
  __builtin_amdgcn_tensor_load_to_lds(g0, g1, z, z, z8, 0);
#else
  __builtin_amdgcn_tensor_load_to_lds(g0, g1, z, z, 0);
#endif
#else
  (void)lds_dst; (void)gsrc; (void)nelem;
#endif
}

__device__ __forceinline__ void tdm_wait() {
#if HAVE_TDM
#if __has_builtin(__builtin_amdgcn_s_wait_tensorcnt)
  __builtin_amdgcn_s_wait_tensorcnt(0);
#else
  asm volatile("s_wait_tensorcnt 0x0" ::: "memory");
#endif
#endif
}

// ---------------------------------------------------------------------------
// WMMA fragment helpers (wave32; A and B both use contiguous k = 16*(lane/16) + e).
// ---------------------------------------------------------------------------
__device__ __forceinline__ v8f wmma32(const v16h& a, const v16h& b, const v8f& c) {
  return __builtin_amdgcn_wmma_f32_16x16x32_f16(
      /*neg_a=*/false, a, /*neg_b=*/false, b,
      /*c_mod=*/(short)0, c, /*reuse_a=*/false, /*reuse_b=*/false);
}

// A from row-major A[m][k]
__device__ __forceinline__ v16h frag_a_rowmajor(const _Float16* src, int ld) {
  int lane = threadIdx.x & 31;
  int m  = lane & 15;
  int kb = (lane >> 4) << 4;            // 0 or 16
  const _Float16* r = src + m * ld;
  v16h f;
#pragma unroll
  for (int e = 0; e < 8; ++e) f[e] = r[kb + e];
#pragma unroll
  for (int e = 0; e < 8; ++e) f[8 + e] = r[kb + 8 + e];
  return f;
}

// A where logical A[m][k] = src[k*ld + m]  (k-major source: A-transposed)
__device__ __forceinline__ v16h frag_a_colmajor(const _Float16* src, int ld) {
  int lane = threadIdx.x & 31;
  int m  = lane & 15;
  int kb = (lane >> 4) << 4;
  v16h f;
#pragma unroll
  for (int e = 0; e < 8; ++e) f[e] = src[(kb + e) * ld + m];
#pragma unroll
  for (int e = 0; e < 8; ++e) f[8 + e] = src[(kb + 8 + e) * ld + m];
  return f;
}

// B[k][n] sourced from row-major B-transposed storage srcT[n][k]
__device__ __forceinline__ v16h frag_b_from_bt(const _Float16* srcT, int ld) {
  int lane = threadIdx.x & 31;
  const _Float16* r = srcT + (lane & 15) * ld + ((lane >> 4) << 4);
  v16h f;
#pragma unroll
  for (int e = 0; e < 16; ++e) f[e] = r[e];
  return f;
}

// Same but from f32 storage (weights stay f32 in global; L2-resident)
__device__ __forceinline__ v16h frag_b_from_bt_f32(const float* srcT, int ld) {
  int lane = threadIdx.x & 31;
  const float* r = srcT + (lane & 15) * ld + ((lane >> 4) << 4);
  v16h f;
#pragma unroll
  for (int e = 0; e < 16; ++e) f[e] = (_Float16)r[e];
  return f;
}

// B from row-major B[k][n]
__device__ __forceinline__ v16h frag_b_rowmajor(const _Float16* src, int ld) {
  int lane = threadIdx.x & 31;
  int n  = lane & 15;
  int kb = (lane >> 4) << 4;
  v16h f;
#pragma unroll
  for (int e = 0; e < 16; ++e) f[e] = src[(kb + e) * ld + n];
  return f;
}

__device__ __forceinline__ void store_c(float* dst, int ld, const v8f& c) {
  int lane = threadIdx.x & 31;
  int n  = lane & 15;
  int mb = (lane >> 4) << 3;
#pragma unroll
  for (int r = 0; r < 8; ++r) dst[(mb + r) * ld + n] = c[r];
}

__device__ __forceinline__ void atomic_add_c(float* dst, int ld, const v8f& c) {
  int lane = threadIdx.x & 31;
  int n  = lane & 15;
  int mb = (lane >> 4) << 3;
#pragma unroll
  for (int r = 0; r < 8; ++r) atomicAdd(&dst[(mb + r) * ld + n], c[r]);
}

// Order-preserving float <-> uint map for atomicMax on floats
__device__ __forceinline__ unsigned f2o(float f) {
  unsigned b = __float_as_uint(f);
  return (b & 0x80000000u) ? ~b : (b | 0x80000000u);
}
__device__ __forceinline__ float o2f(unsigned o) {
  unsigned b = (o & 0x80000000u) ? (o & 0x7FFFFFFFu) : ~o;
  return __uint_as_float(b);
}

__device__ __forceinline__ int wave_id() {
  return __builtin_amdgcn_readfirstlane((int)(threadIdx.x >> 5));
}

// ---------------------------------------------------------------------------
// Kernel 0: zero context / k_sum, init global k-max scalar
// ---------------------------------------------------------------------------
__global__ void init_kernel(float* ctx, float* ksum, unsigned* kmax) {
  int i = blockIdx.x * 256 + threadIdx.x;
  if (i < NB * 128 * 128) ctx[i] = 0.0f;
  if (i < NB * 128)       ksum[i] = 0.0f;
  if (i == 0)             *kmax = f2o(-3.0e38f);
}

// ---------------------------------------------------------------------------
// Kernel 1: fused 1x1 convs (w1|w2|wa) as GEMM [32 px x 128 K x 256 out]
//           + channel L2 norm + Performer diag precompute.
// ---------------------------------------------------------------------------
__global__ void conv_kernel(const float* __restrict__ x,
                            const float* __restrict__ w1, const float* __restrict__ b1,
                            const float* __restrict__ w2, const float* __restrict__ b2,
                            const float* __restrict__ wa, const float* __restrict__ ba,
                            _Float16* __restrict__ q, _Float16* __restrict__ k,
                            _Float16* __restrict__ v,
                            float* __restrict__ diagq, float* __restrict__ diagk) {
  __shared__ union ConvS {
    _Float16 xs[32 * 128];   //  8 KB: x tile, pixel-major, live during GEMM
    float    res[32 * 256];  // 32 KB: conv result, live after GEMM
  } u;
  __shared__ float s1S[32], s2S[32];

  const long basePix = (long)blockIdx.x * 32;
  const int  batch = (int)(basePix >> 14);
  const int  hw0   = (int)(basePix & (LLEN - 1));
  const int  t     = threadIdx.x;

  // x[n][c][hw] -> xs[p][c] (f16). float4 loads: rows are 16-B aligned
  // (basePix is 32-aligned). Contiguous pixels per channel => coalesced.
  const float* xb = x + ((long)batch * CC) * LLEN + hw0;
  for (int i = t; i < 32 * 128 / 4; i += 256) {     // 1024 float4 tasks
    int c = i >> 3, p4 = (i & 7) << 2;
    float4 f = *(const float4*)(xb + (long)c * LLEN + p4);
    u.xs[(p4 + 0) * 128 + c] = (_Float16)f.x;
    u.xs[(p4 + 1) * 128 + c] = (_Float16)f.y;
    u.xs[(p4 + 2) * 128 + c] = (_Float16)f.z;
    u.xs[(p4 + 3) * 128 + c] = (_Float16)f.w;
  }
  __syncthreads();

  const int wave   = wave_id();                 // SGPR (readfirstlane)
  const int pixRow = wave >> 2;   // 0..1 (16 pixels each)
  const int colQ   = wave & 3;    // 0..3 (64 output channels each)

  v8f acc[4] = {};
  for (int ks = 0; ks < 4; ++ks) {              // K = 128 = 4 x 32
    v16h a = frag_a_rowmajor(u.xs + pixRow * 16 * 128 + ks * 32, 128);
#pragma unroll
    for (int j = 0; j < 4; ++j) {
      // o0 is wave-uniform (colQ from readfirstlane, j unrolled constant):
      // scalar branches, weight base stays in SGPRs -> global_load (not flat).
      int o0 = colQ * 64 + j * 16;
      const float* wr;
      if (o0 < 64)        wr = w1 + o0 * 128;
      else if (o0 < 128)  wr = w2 + (o0 - 64) * 128;
      else                wr = wa + (o0 - 128) * 128;
      v16h b = frag_b_from_bt_f32(wr + ks * 32, 128);
      acc[j] = wmma32(a, b, acc[j]);
    }
  }
  __syncthreads();                 // all reads of u.xs done before res overwrite
#pragma unroll
  for (int j = 0; j < 4; ++j)
    store_c(u.res + pixRow * 16 * 256 + (colQ * 64 + j * 16), 256, acc[j]);
  __syncthreads();

  // bias add
  for (int i = t; i < 32 * 256; i += 256) {
    int o = i & 255;
    float bb = (o < 64) ? b1[o] : (o < 128 ? b2[o - 64] : ba[o - 128]);
    u.res[i] += bb;
  }
  __syncthreads();

  // per-pixel L2 norms over e1 (0..63) and e2 (64..127), diag precompute
  if (t < 32) {
    float s1 = 0.f, s2 = 0.f;
    for (int d = 0; d < 64; ++d) { float a1 = u.res[t * 256 + d];       s1 += a1 * a1; }
    for (int d = 0; d < 64; ++d) { float a2 = u.res[t * 256 + 64 + d];  s2 += a2 * a2; }
    float c1 = (K_SCALE * DNORM) / fmaxf(sqrtf(s1), NORM_EPS);
    float c2 = (K_SCALE * DNORM) / fmaxf(sqrtf(s2), NORM_EPS);
    s1S[t] = c1; s2S[t] = c2;
    diagq[basePix + t] = 0.5f * c1 * c1 * s1;
    diagk[basePix + t] = 0.5f * c2 * c2 * s2;
  }
  __syncthreads();

  for (int i = t; i < 32 * 256; i += 256) {
    int p = i >> 8, o = i & 255;
    float val = u.res[i];
    if (o < 64)        q[(basePix + p) * 64 + o]          = (_Float16)(val * s1S[p]);
    else if (o < 128)  k[(basePix + p) * 64 + (o - 64)]   = (_Float16)(val * s2S[p]);
    else               v[(basePix + p) * 128 + (o - 128)] = (_Float16)val;
  }
}

// ---------------------------------------------------------------------------
// Shared body: data_dash = src(64 px x 64) @ proj^T (64 x 128) -> ddS (f32).
// inS staged by TDM (wave 0) overlapped with projS f32->f16 conversion.
// ---------------------------------------------------------------------------
__device__ void feature_gemm_body(const _Float16* __restrict__ src,
                                  const float* __restrict__ proj,
                                  _Float16* inS /*64*64*/, _Float16* projS /*128*64*/,
                                  float* ddS /*64*128*/, long basePix) {
  int t = threadIdx.x;
  int wave = wave_id();
#if HAVE_TDM
  if (wave == 0) tdm_copy_f16(inS, src + basePix * 64, 64 * 64);
#endif
  for (int i = t; i < 128 * 64; i += 256) projS[i] = (_Float16)proj[i];
#if HAVE_TDM
  if (wave == 0) tdm_wait();
#else
  for (int i = t; i < 64 * 64; i += 256) inS[i] = src[basePix * 64 + i];
#endif
  __syncthreads();

  int pixRow = wave >> 1;          // 0..3 (16 pixels each)
  int fHalf  = wave & 1;           // 0..1 (64 features each)
  v8f acc[4] = {};
  for (int ks = 0; ks < 2; ++ks) { // K = 64 = 2 x 32
    v16h a = frag_a_rowmajor(inS + pixRow * 16 * 64 + ks * 32, 64);
#pragma unroll
    for (int j = 0; j < 4; ++j) {
      int m0 = fHalf * 64 + j * 16;
      v16h b = frag_b_from_bt(projS + m0 * 64 + ks * 32, 64);  // B[k][n]=proj[n][k]
      acc[j] = wmma32(a, b, acc[j]);
    }
  }
#pragma unroll
  for (int j = 0; j < 4; ++j)
    store_c(ddS + pixRow * 16 * 128 + (fHalf * 64 + j * 16), 128, acc[j]);
  __syncthreads();
}

// ---------------------------------------------------------------------------
// Kernel 2: qp = ratio * (exp(dd - diag - rowmax) + eps)
// ---------------------------------------------------------------------------
__global__ void qp_kernel(const _Float16* __restrict__ q, const float* __restrict__ proj,
                          const float* __restrict__ diagq, _Float16* __restrict__ qp) {
  __shared__ _Float16 inS[64 * 64];
  __shared__ _Float16 projS[128 * 64];
  __shared__ float    ddS[64 * 128];
  __shared__ float    shiftS[64];
  long basePix = (long)blockIdx.x * 64;
  feature_gemm_body(q, proj, inS, projS, ddS, basePix);
  int t = threadIdx.x;
  if (t < 64) {
    float mx = -3.0e38f;
    for (int m = 0; m < 128; ++m) mx = fmaxf(mx, ddS[t * 128 + m]);
    shiftS[t] = diagq[basePix + t] + mx;
  }
  __syncthreads();
  for (int i = t; i < 64 * 128; i += 256) {
    int p = i >> 7;
    float val = RATIO * (__expf(ddS[i] - shiftS[p]) + KERN_EPS);
    qp[basePix * 128 + i] = (_Float16)val;
  }
}

// ---------------------------------------------------------------------------
// Kernel 3: global max of k's data_dash (Performer key stabilizer)
// ---------------------------------------------------------------------------
__global__ void kmax_kernel(const _Float16* __restrict__ k, const float* __restrict__ proj,
                            unsigned* __restrict__ kmax) {
  __shared__ _Float16 inS[64 * 64];
  __shared__ _Float16 projS[128 * 64];
  __shared__ float    ddS[64 * 128];
  __shared__ float    red[256];
  long basePix = (long)blockIdx.x * 64;
  feature_gemm_body(k, proj, inS, projS, ddS, basePix);
  int t = threadIdx.x;
  float mx = -3.0e38f;
  for (int i = t; i < 64 * 128; i += 256) mx = fmaxf(mx, ddS[i]);
  red[t] = mx;
  __syncthreads();
  for (int s = 128; s > 0; s >>= 1) {
    if (t < s) red[t] = fmaxf(red[t], red[t + s]);
    __syncthreads();
  }
  if (t == 0) atomicMax(kmax, f2o(red[0]));
}

// ---------------------------------------------------------------------------
// Kernel 4: kp = ratio * (exp(dd - diag - globalmax) + eps); accumulate k_sum
// ---------------------------------------------------------------------------
__global__ void kp_kernel(const _Float16* __restrict__ k, const float* __restrict__ proj,
                          const float* __restrict__ diagk, const unsigned* __restrict__ kmax,
                          _Float16* __restrict__ kp, float* __restrict__ ksum) {
  __shared__ _Float16 inS[64 * 64];
  __shared__ _Float16 projS[128 * 64];
  __shared__ float    ddS[64 * 128];
  __shared__ float    shiftS[64];
  long basePix = (long)blockIdx.x * 64;
  feature_gemm_body(k, proj, inS, projS, ddS, basePix);
  int t = threadIdx.x;
  float gmax = o2f(*kmax);
  if (t < 64) shiftS[t] = diagk[basePix + t] + gmax;
  __syncthreads();
  for (int i = t; i < 64 * 128; i += 256) {
    int p = i >> 7;
    float val = RATIO * (__expf(ddS[i] - shiftS[p]) + KERN_EPS);
    kp[basePix * 128 + i] = (_Float16)val;
    ddS[i] = val;
  }
  __syncthreads();
  if (t < 128) {
    float s = 0.f;
    for (int p = 0; p < 64; ++p) s += ddS[p * 128 + t];
    int batch = (int)(basePix >> 14);
    atomicAdd(&ksum[batch * 128 + t], s);
  }
}

// ---------------------------------------------------------------------------
// Kernel 5: context += kp_chunk^T (128 x 64) @ v_chunk (64 x 128), split-K.
// kp/v tiles staged by TDM.
// ---------------------------------------------------------------------------
__global__ void ctx_kernel(const _Float16* __restrict__ kp, const _Float16* __restrict__ v,
                           float* __restrict__ ctx) {
  __shared__ _Float16 kpS[64 * 128];  // 16 KB  [l][m]
  __shared__ _Float16 vS[64 * 128];   // 16 KB  [l][e]
  long base = (long)blockIdx.x * 64;
  int batch = (int)(base >> 14);
  int t = threadIdx.x;
  int wave = wave_id();
#if HAVE_TDM
  if (wave == 0) {
    tdm_copy_f16(kpS, kp + base * 128, 64 * 128);
    tdm_copy_f16(vS,  v  + base * 128, 64 * 128);
    tdm_wait();
  }
#else
  for (int i = t; i < 64 * 128; i += 256) {
    kpS[i] = kp[base * 128 + i];
    vS[i]  = v[base * 128 + i];
  }
#endif
  __syncthreads();

  int m0 = wave * 16;                 // 8 waves cover all 128 features
  v8f acc[8] = {};
  for (int ks = 0; ks < 2; ++ks) {    // K = 64 pixels = 2 x 32
    // A = kp^T: A[m][kk] = kpS[kk][m]  -> k-major gather
    v16h a = frag_a_colmajor(kpS + ks * 32 * 128 + m0, 128);
#pragma unroll
    for (int j = 0; j < 8; ++j) {
      v16h b = frag_b_rowmajor(vS + ks * 32 * 128 + j * 16, 128);
      acc[j] = wmma32(a, b, acc[j]);
    }
  }
  float* cbase = ctx + (long)batch * 128 * 128 + m0 * 128;
#pragma unroll
  for (int j = 0; j < 8; ++j) atomic_add_c(cbase + j * 16, 128, acc[j]);
  (void)t;
}

// ---------------------------------------------------------------------------
// Kernel 6: out = (qp @ ctx) * (1/(qp . ksum)) * 0.1 + x   (NCHW f32 output).
// qp tile staged by TDM overlapped with ctx f32->f16 conversion.
// ---------------------------------------------------------------------------
__global__ void out_kernel(const _Float16* __restrict__ qp, const float* __restrict__ ctx,
                           const float* __restrict__ ksum, const float* __restrict__ x,
                           float* __restrict__ out) {
  __shared__ _Float16 qpS[64 * 128];  // 16 KB
  __shared__ union OutS {
    _Float16 ctxS[128 * 128];         // 32 KB, live during GEMM
    float    resS[64 * 128];          // 32 KB, live after GEMM
  } u;
  __shared__ float dinvS[64];
  __shared__ float ksumS[128];

  long basePix = (long)blockIdx.x * 64;
  int batch = (int)(basePix >> 14);
  int hw0   = (int)(basePix & (LLEN - 1));
  int t = threadIdx.x;
  int wave = wave_id();

#if HAVE_TDM
  if (wave == 0) tdm_copy_f16(qpS, qp + basePix * 128, 64 * 128);
#else
  for (int i = t; i < 64 * 128; i += 256) qpS[i] = qp[basePix * 128 + i];
#endif
  const float* cg = ctx + (long)batch * 128 * 128;
  for (int i = t; i < 128 * 128; i += 256) {
    __builtin_prefetch(cg + i + 4096, 0, 0);   // global_prefetch_b8
    u.ctxS[i] = (_Float16)cg[i];
  }
  if (t < 128) ksumS[t] = ksum[batch * 128 + t];
#if HAVE_TDM
  if (wave == 0) tdm_wait();
#endif
  __syncthreads();

  int pixRow = wave >> 1;             // 0..3
  int cHalf  = wave & 1;              // 0..1
  v8f acc[4] = {};
  for (int ks = 0; ks < 4; ++ks) {    // K = 128 features = 4 x 32
    v16h a = frag_a_rowmajor(qpS + pixRow * 16 * 128 + ks * 32, 128);
#pragma unroll
    for (int j = 0; j < 4; ++j) {
      int e0 = cHalf * 64 + j * 16;
      v16h b = frag_b_rowmajor(u.ctxS + ks * 32 * 128 + e0, 128);  // B[k][n]=ctx[k][n]
      acc[j] = wmma32(a, b, acc[j]);
    }
  }
  __syncthreads();                    // all ctxS reads done before res overwrite
#pragma unroll
  for (int j = 0; j < 4; ++j)
    store_c(u.resS + pixRow * 16 * 128 + (cHalf * 64 + j * 16), 128, acc[j]);
  __syncthreads();

  if (t < 64) {
    float den = 0.f;
    for (int m = 0; m < 128; ++m) den += (float)qpS[t * 128 + m] * ksumS[m];
    dinvS[t] = 1.0f / den;
  }
  __syncthreads();

  for (int i = t; i < 64 * 128; i += 256) {
    int c = i >> 6, p = i & 63;       // fixed channel, contiguous pixels: coalesced
    long gi = ((long)batch * CC + c) * LLEN + hw0 + p;
    out[gi] = u.resS[p * 128 + c] * dinvS[p] * RES_SCALE + x[gi];
  }
}

// ---------------------------------------------------------------------------
extern "C" void kernel_launch(void* const* d_in, const int* in_sizes, int n_in,
                              void* d_out, int out_size, void* d_ws, size_t ws_size,
                              hipStream_t stream) {
  (void)in_sizes; (void)n_in; (void)out_size; (void)ws_size;
  const float* x    = (const float*)d_in[0];
  const float* w1   = (const float*)d_in[1];
  const float* b1   = (const float*)d_in[2];
  const float* w2   = (const float*)d_in[3];
  const float* b2   = (const float*)d_in[4];
  const float* wa   = (const float*)d_in[5];
  const float* ba   = (const float*)d_in[6];
  const float* proj = (const float*)d_in[7];
  float* out = (float*)d_out;

  char* ws = (char*)d_ws;
  _Float16* q  = (_Float16*)ws;                         // NL*64  f16  (16.8 MB)
  _Float16* k  = q  + (size_t)NL * 64;                  // NL*64  f16
  _Float16* v  = k  + (size_t)NL * 64;                  // NL*128 f16  (33.6 MB)
  _Float16* qp = v  + (size_t)NL * 128;                 // NL*128 f16
  _Float16* kp = qp + (size_t)NL * 128;                 // NL*128 f16
  float* diagq = (float*)(kp + (size_t)NL * 128);       // NL f32
  float* diagk = diagq + NL;                            // NL f32
  float* ksum  = diagk + NL;                            // NB*128 f32
  float* ctx   = ksum + NB * 128;                       // NB*128*128 f32
  unsigned* kmax = (unsigned*)(ctx + NB * 128 * 128);   // 1 u32

  dim3 blk(256);
  init_kernel<<<dim3((NB * 128 * 128 + 255) / 256), blk, 0, stream>>>(ctx, ksum, kmax);
  conv_kernel<<<dim3(NL / 32), blk, 0, stream>>>(x, w1, b1, w2, b2, wa, ba,
                                                 q, k, v, diagq, diagk);
  qp_kernel  <<<dim3(NL / 64), blk, 0, stream>>>(q, proj, diagq, qp);
  kmax_kernel<<<dim3(NL / 64), blk, 0, stream>>>(k, proj, kmax);
  kp_kernel  <<<dim3(NL / 64), blk, 0, stream>>>(k, proj, diagk, kmax, kp, ksum);
  ctx_kernel <<<dim3(NL / 64), blk, 0, stream>>>(kp, v, ctx);
  out_kernel <<<dim3(NL / 64), blk, 0, stream>>>(qp, ctx, ksum, x, out);
}
